// ZynqLinear_38079180047085
// MI455X (gfx1250) — matrix operand order repaired and checked by reading, not compiled, not run on hardware
//
#include <hip/hip_runtime.h>
#include <cstdint>

// ---------------------------------------------------------------------------
// ZynqLinear: out[8192,2048] = x[8192,2048](f32) . W[2048,2048](int8)^T + bias
//
// MI455X reasoning:
//  * Working set (x 64MB + W 4MB + out 64MB) fits the 192MB L2 -> after first
//    touch this is WMMA-throughput bound, not HBM bound (AI ~520 FLOP/B).
//  * int8 weights are exact in f16; f32->f16 activations is the standard
//    W8A16 trade. V_WMMA_F32_16X16X32_F16 with f32 accumulate (like the ref).
//  * Wave tile 32x64 (2 M-subtiles x 4 N-tiles): 16 b128 loads feed 8 WMMAs
//    per 32-K step; each B fragment feeds two independent WMMAs.
//  * K loop: outer step 128, inner 4x32 fully unrolled so K displacement
//    folds into the 24-bit load offsets (pointer bumps 1x/128K, away from
//    the WMMAs -> fewer WMMA->VALU hazard v_nops).
//  * W is [out,in] row-major: each output feature's K-run is contiguous,
//    exactly the per-lane layout the WMMA B-matrix wants. No transpose.
//  * Pre-pass dequantizes W -> f16 in d_ws (8MB): zero int8 unpack in loop.
// ---------------------------------------------------------------------------

typedef __attribute__((ext_vector_type(16))) _Float16 v16h;
typedef __attribute__((ext_vector_type(8)))  _Float16 v8h;
typedef __attribute__((ext_vector_type(4)))  _Float16 v4h;
typedef __attribute__((ext_vector_type(8)))  float    v8f;
typedef __attribute__((ext_vector_type(4)))  float    v4f;

#define M_DIM 8192
#define K_DIM 2048
#define N_DIM 2048
#define BM    128     // block M tile: 4 wave-rows x 32
#define BN    128     // block N tile: 2 wave-cols x 64
#define WM    32      // wave M tile (2 wmma tiles)
#define WN    64      // wave N tile (4 wmma tiles)
#define KG    128     // K group per outer iteration (4 x 32 unrolled)

// ---------------------------------------------------------------------------
// Pre-pass: int8 weight -> f16 weight in workspace. 8 elements per thread.
// ---------------------------------------------------------------------------
__global__ __launch_bounds__(256) void zynq_w8_to_f16(
    const int8_t* __restrict__ w8, _Float16* __restrict__ wf, int n)
{
    int i = (blockIdx.x * blockDim.x + threadIdx.x) * 8;
    if (i >= n) return;
    int2 raw = *(const int2*)(w8 + i);   // 8-byte aligned (i % 8 == 0)
    v8h h;
#pragma unroll
    for (int t = 0; t < 4; ++t)
        h[t] = (_Float16)(float)(int8_t)(raw.x >> (8 * t));
#pragma unroll
    for (int t = 0; t < 4; ++t)
        h[4 + t] = (_Float16)(float)(int8_t)(raw.y >> (8 * t));
    *(v8h*)(wf + i) = h;
}

// Load one 16x32 f16 A-fragment from an f32 row stream (pre-offset per lane).
// ISA layout: lanes 0-15 carry K = {0..7, 16..23}, lanes 16-31 carry
// K = {8..15, 24..31} of row (lane&15).
__device__ __forceinline__ v16h load_a_frag(const float* __restrict__ ap)
{
    v4f f0 = *(const v4f*)(ap);
    v4f f1 = *(const v4f*)(ap + 4);
    v4f f2 = *(const v4f*)(ap + 16);
    v4f f3 = *(const v4f*)(ap + 20);
    v4h h0 = __builtin_convertvector(f0, v4h);
    v4h h1 = __builtin_convertvector(f1, v4h);
    v4h h2 = __builtin_convertvector(f2, v4h);
    v4h h3 = __builtin_convertvector(f3, v4h);
    v8h lo = __builtin_shufflevector(h0, h1, 0, 1, 2, 3, 4, 5, 6, 7);
    v8h hi = __builtin_shufflevector(h2, h3, 0, 1, 2, 3, 4, 5, 6, 7);
    return __builtin_shufflevector(lo, hi, 0, 1, 2, 3, 4, 5, 6, 7,
                                   8, 9, 10, 11, 12, 13, 14, 15);
}

// Load one 32x16 f16 B-fragment (pre-offset per lane; contiguous K run).
__device__ __forceinline__ v16h load_b_frag(const _Float16* __restrict__ bp)
{
    v8h lo = *(const v8h*)(bp);
    v8h hi = *(const v8h*)(bp + 16);
    return __builtin_shufflevector(lo, hi, 0, 1, 2, 3, 4, 5, 6, 7,
                                   8, 9, 10, 11, 12, 13, 14, 15);
}

// ---------------------------------------------------------------------------
// Main GEMM: 256 threads = 8 wave32s arranged 4(M) x 2(N).
// Wave computes a 32x64 strip: acc[2][4] 16x16 tiles.
// ---------------------------------------------------------------------------
__global__ __launch_bounds__(256) void zynq_linear_wmma(
    const float*    __restrict__ x,
    const _Float16* __restrict__ wf,
    const int*      __restrict__ bias,
    float*          __restrict__ out)
{
    const int lane = threadIdx.x & 31;
    const int wave = threadIdx.x >> 5;
    const int l16  = lane & 15;
    const int hi   = lane >> 4;                 // lane half
    const int bm   = blockIdx.y * BM + (wave >> 1) * WM;
    const int bn   = blockIdx.x * BN + (wave & 1) * WN;

    // A: per-lane row pointers for the two 16-row subtiles.
    const int klo = hi << 3;                    // 0 or 8
    const float* xrow0 = x + (long)(bm + l16) * K_DIM + klo;
    const float* xrow1 = xrow0 + (long)16 * K_DIM;

    // B (16-bit 32x16 layout): lane holds column bn+j*16+l16; lanes 0-15
    // carry K={0..7,16..23}, lanes 16-31 carry K={8..15,24..31}.
    const _Float16* wcol[4];
#pragma unroll
    for (int j = 0; j < 4; ++j)
        wcol[j] = wf + (long)(bn + j * 16 + l16) * K_DIM + (hi << 3);

    v8f acc[2][4];
#pragma unroll
    for (int m = 0; m < 2; ++m)
#pragma unroll
        for (int j = 0; j < 4; ++j)
            acc[m][j] = v8f{};

    for (int kk0 = 0; kk0 < K_DIM; kk0 += KG) {
        // One speculative prefetch per stream per K group (global_prefetch_b8).
        __builtin_prefetch(xrow0 + kk0 + KG, 0, 1);
        __builtin_prefetch(xrow1 + kk0 + KG, 0, 1);
        __builtin_prefetch(wcol[0] + kk0 + KG, 0, 1);

#pragma unroll
        for (int ku = 0; ku < KG; ku += 32) {
            const int kk = kk0 + ku;
            v16h a0 = load_a_frag(xrow0 + kk);
            v16h a1 = load_a_frag(xrow1 + kk);

#pragma unroll
            for (int j = 0; j < 4; ++j) {
                v16h b = load_b_frag(wcol[j] + kk);
                acc[0][j] = __builtin_amdgcn_wmma_f32_16x16x32_f16(
                    false, a0, false, b, (short)0, acc[0][j], false, false);
                acc[1][j] = __builtin_amdgcn_wmma_f32_16x16x32_f16(
                    false, a1, false, b, (short)0, acc[1][j], false, false);
            }
        }
    }

    // ---- Epilogue: C/D 16x16 f32 layout. VGPR v: row = v (lanes 0-15) or
    // v+8 (lanes 16-31); column = lane&15. Add bias, store. ----
#pragma unroll
    for (int m = 0; m < 2; ++m) {
        const int crow = bm + m * 16 + (hi << 3);
#pragma unroll
        for (int j = 0; j < 4; ++j) {
            const int col = bn + j * 16 + l16;
            const float bf = (float)bias[col];
            float* op = out + (long)crow * N_DIM + col;
#pragma unroll
            for (int v = 0; v < 8; ++v)
                op[(long)v * N_DIM] = acc[m][j][v] + bf;
        }
    }
}

// ---------------------------------------------------------------------------
extern "C" void kernel_launch(void* const* d_in, const int* in_sizes, int n_in,
                              void* d_out, int out_size, void* d_ws, size_t ws_size,
                              hipStream_t stream)
{
    const float*  x    = (const float*)d_in[0];
    const int8_t* w8   = (const int8_t*)d_in[1];
    const int*    bias = (const int*)d_in[2];
    float*        out  = (float*)d_out;
    _Float16*     wf   = (_Float16*)d_ws;   // 2048*2048*2 = 8 MB scratch

    // 1) Dequantize weights once (int8 -> f16), 8 elems/thread.
    const int wn = N_DIM * K_DIM;                        // 4,194,304
    zynq_w8_to_f16<<<(wn / 8) / 256, 256, 0, stream>>>(w8, wf, wn);

    // 2) WMMA GEMM: grid (N/BN, M/BM) = (16, 64), 256 threads (8 wave32s).
    dim3 grid(N_DIM / BN, M_DIM / BM);
    zynq_linear_wmma<<<grid, 256, 0, stream>>>(x, wf, bias, out);
}
